// DyIntraModalityUpdate_76270029242468
// MI455X (gfx1250) — matrix operand order repaired and checked by reading, not compiled, not run on hardware
//
#include <hip/hip_runtime.h>
#include <hip/hip_bf16.h>
#include <cstdint>

typedef __attribute__((ext_vector_type(16))) __bf16        v16bf;
typedef __attribute__((ext_vector_type(8)))  float         v8f;
typedef __attribute__((ext_vector_type(4)))  unsigned int  u32x4;

namespace {

constexpr int Bb     = 64;
constexpr int NOBJ   = 768;
constexpr int MAXLEN = 256;
constexpr int DM     = 512;   // model dim
constexpr int H      = 8;
constexpr int DH     = 64;

// WMMA operand container: v16bf for the builtin, vector-load/packing views.
union ABf {
  v16bf v;
  __bf16 h[16];
  unsigned short s[16];
  u32x4 u4[2];
};

__device__ __forceinline__ unsigned short bf(float f) {
  union { __bf16 b; unsigned short s; } u;
  u.b = (__bf16)f;
  return u.s;
}

__device__ __forceinline__ v8f wmma_bf16(const ABf& a, const ABf& b, v8f c) {
  return __builtin_amdgcn_wmma_f32_16x16x32_bf16(
      /*neg_a=*/false, a.v, /*neg_b=*/false, b.v,
      /*c_mod=*/(short)0, c, /*reuse_a=*/false, /*reuse_b=*/false);
}

// ---------------------------------------------------------------- small prep
__global__ void mean_kernel(const float* __restrict__ v,
                            const float* __restrict__ vmask,
                            float* __restrict__ vmean) {
  const int b = blockIdx.x;
  const int t = threadIdx.x;                 // 256 threads
  float msum = 0.f;
  for (int n = 0; n < NOBJ; ++n) msum += vmask[b * NOBJ + n];
  const float inv = 1.f / msum;
  for (int kk = 0; kk < 2; ++kk) {
    const int k = t + kk * 256;
    float acc = 0.f;
    for (int n = 0; n < NOBJ; ++n)
      acc += v[((size_t)b * NOBJ + n) * DM + k] * vmask[b * NOBJ + n];
    vmean[b * DM + k] = acc * inv;
  }
}

// gq gates q-attention: 1+sigmoid(v_mean@Wv4q.T+bv4q)
// gv gates v-attention: 1+sigmoid(q_trans@Wq4v.T+bq4v)
__global__ void gate_kernel(const float* __restrict__ vmean,
                            const float* __restrict__ qtrans,
                            const float* __restrict__ Wv4q, const float* __restrict__ bv4q,
                            const float* __restrict__ Wq4v, const float* __restrict__ bq4v,
                            float* __restrict__ gq, float* __restrict__ gv) {
  const int b = blockIdx.x;
  const int t = threadIdx.x;
  for (int jj = 0; jj < 2; ++jj) {
    const int j = t + jj * 256;
    float a0 = bv4q[j], a1 = bq4v[j];
    for (int k = 0; k < DM; ++k) {
      a0 += vmean[b * DM + k]  * Wv4q[j * DM + k];
      a1 += qtrans[b * DM + k] * Wq4v[j * DM + k];
    }
    gq[b * DM + j] = 1.f + 1.f / (1.f + __expf(-a0));
    gv[b * DM + j] = 1.f + 1.f / (1.f + __expf(-a1));
  }
}

// 8 elements per thread: two b128 loads, one b128 store
__global__ void cvt_bf16_kernel(const float* __restrict__ src,
                                unsigned short* __restrict__ dst, int n) {
  const int i = (blockIdx.x * blockDim.x + threadIdx.x) * 8;
  if (i >= n) return;
  union { u32x4 u; __bf16 h[8]; } o;
#pragma unroll
  for (int k = 0; k < 8; ++k) o.h[k] = (__bf16)src[i + k];
  *(u32x4*)(dst + i) = o.u;
}

// residual add + convert: dst = bf16(X + U)
__global__ void addcvt_kernel(const float* __restrict__ X, const float* __restrict__ U,
                              unsigned short* __restrict__ dst, int n) {
  const int i = (blockIdx.x * blockDim.x + threadIdx.x) * 8;
  if (i >= n) return;
  union { u32x4 u; __bf16 h[8]; } o;
#pragma unroll
  for (int k = 0; k < 8; ++k) o.h[k] = (__bf16)(X[i + k] + U[i + k]);
  *(u32x4*)(dst + i) = o.u;
}

// ------------------------------------------------- fused QKV transform GEMM
// 64x64 output block per wave: 4 A-operands x 4 B-operands, 16 accumulators.
// out[row,j] = (X16[row,:]@W[j,:] + bias[j]) * mask[row] * gate[b, j%512]
// j <  1024 -> KQ[row*1024 + j]            (key | query, row-major)
// j >= 1024 -> VT[(b*512 + j-1024)*T + n]  (value, transposed per batch)
template <int T>
__global__ void __launch_bounds__(256)
xform_kernel(const unsigned short* __restrict__ X16,
             const unsigned short* __restrict__ W16,
             const float* __restrict__ bias, const float* __restrict__ mask,
             const float* __restrict__ gate, unsigned short* __restrict__ KQ,
             unsigned short* __restrict__ VT) {
  const int lane = threadIdx.x & 31;
  const int wave = threadIdx.x >> 5;
  const int tileId = blockIdx.x * 8 + wave;
  const int Ntiles = (3 * DM) / 64;          // 24
  const int tm = tileId / Ntiles, tn = tileId % Ntiles;

  const int am  = lane & 15;                 // A: row within 16-tile
  const int akb = (lane >> 4) * 16;          // A: K sub-offset
  const int bn  = lane & 15;                 // B: col within 16-tile
  const int bkb = (lane >> 4) * 16;          // B: K sub-offset
  const int rm  = tm * 64;
  const int jn  = tn * 64;

  const unsigned short* xbase = X16 + ((size_t)(rm + am)) * DM;       // +i*16*DM
  const unsigned short* wbase = W16 + ((size_t)(jn + bn)) * DM;       // +g*16*DM

  v8f acc[4][4] = {};
  for (int kc = 0; kc < DM / 32; ++kc) {
    const int k0 = kc * 32 + akb;
    const int k1 = kc * 32 + bkb;
    ABf a[4], b[4];
#pragma unroll
    for (int i = 0; i < 4; ++i) {
      const unsigned short* xr = xbase + (size_t)i * 16 * DM;
      a[i].u4[0] = *(const u32x4*)(xr + k0);
      a[i].u4[1] = *(const u32x4*)(xr + k0 + 8);
    }
#pragma unroll
    for (int g = 0; g < 4; ++g) {
      const unsigned short* wr = wbase + (size_t)g * 16 * DM;
      b[g].u4[0] = *(const u32x4*)(wr + k1);
      b[g].u4[1] = *(const u32x4*)(wr + k1 + 8);
    }
#pragma unroll
    for (int i = 0; i < 4; ++i)
#pragma unroll
      for (int g = 0; g < 4; ++g)
        acc[i][g] = wmma_bf16(a[i], b[g], acc[i][g]);
  }

  const int cm = (lane >> 4) * 8;
  const int cn = lane & 15;
#pragma unroll
  for (int g = 0; g < 4; ++g) {
    const int jc = jn + g * 16 + cn;
    const float bj = bias[jc];
    const int gidx = jc & (DM - 1);
#pragma unroll
    for (int i = 0; i < 4; ++i) {
#pragma unroll
      for (int r = 0; r < 8; ++r) {
        const int rr = rm + i * 16 + cm + r;
        const int b  = rr / T;               // compile-time T: mul-shift
        const int nn = rr - b * T;
        const float val = (acc[i][g][r] + bj) * mask[rr] * gate[b * DM + gidx];
        const unsigned short o = bf(val);
        if (jc < 2 * DM) KQ[(size_t)rr * 1024 + jc] = o;
        else             VT[((size_t)b * DM + (jc - 2 * DM)) * T + nn] = o;
      }
    }
  }
}

// ----------------------------------------------------------------- attention
// One wave per (b, h, 16-query strip). Scores strip lives in LDS (16*T fp32),
// softmax with wave shuffles, probs re-written in place as bf16, attn@V via
// WMMA using the pre-transposed V. Head H-1 probs scattered (transposed) to out.
template <int T>
__global__ void __launch_bounds__(32)
attn_kernel(const unsigned short* __restrict__ KQ,
            const unsigned short* __restrict__ VT,
            const float* __restrict__ mask,
            float* __restrict__ upd, float* __restrict__ attn_out) {
  __shared__ float smem[16 * T];
  const int lane = threadIdx.x;
  int id = blockIdx.x;
  constexpr int QT = T / 16;
  const int qt = id % QT; id /= QT;
  const int h  = id % H;
  const int b  = id / H;

  const unsigned short* KQb = KQ + (size_t)b * T * 1024;
  const int am  = lane & 15;
  const int akb = (lane >> 4) * 16;
  const int bn  = lane & 15;
  const int bkb = (lane >> 4) * 16;
  const int cm  = (lane >> 4) * 8;
  const int cn  = lane & 15;

  // Q strip (16 x 64) as two A operands (K=0..31, 32..63), gated already
  ABf aq0, aq1;
  {
    const unsigned short* qr = KQb + ((size_t)(qt * 16 + am)) * 1024 + DM + h * DH;
    aq0.u4[0] = *(const u32x4*)(qr + akb);
    aq0.u4[1] = *(const u32x4*)(qr + 8 + akb);
    aq1.u4[0] = *(const u32x4*)(qr + 32 + akb);
    aq1.u4[1] = *(const u32x4*)(qr + 40 + akb);
  }
  // ---- scores = Q K^T / 8, key-masked, to LDS
  for (int kt = 0; kt < QT; ++kt) {
    const unsigned short* kr = KQb + ((size_t)(kt * 16 + bn)) * 1024 + h * DH;
    ABf b0, b1;
    b0.u4[0] = *(const u32x4*)(kr + bkb);
    b0.u4[1] = *(const u32x4*)(kr + bkb + 8);
    b1.u4[0] = *(const u32x4*)(kr + 32 + bkb);
    b1.u4[1] = *(const u32x4*)(kr + 32 + bkb + 8);
    v8f acc = {};
    acc = wmma_bf16(aq0, b0, acc);
    acc = wmma_bf16(aq1, b1, acc);
    const int keyc = kt * 16 + cn;
    const float km = mask[b * T + keyc];
#pragma unroll
    for (int r = 0; r < 8; ++r) {
      float s = acc[r] * 0.125f;
      if (km == 0.f) s = -1e30f;
      smem[(cm + r) * T + keyc] = s;
    }
  }
  __syncthreads();

  // ---- softmax per row; probs stored bf16 aliased over scores
  unsigned short* probs = (unsigned short*)smem;
  constexpr int CPL = T / 32;
  const bool emit_attn = (attn_out != nullptr) && (h == H - 1);
  for (int r = 0; r < 16; ++r) {
    float vals[CPL];
#pragma unroll
    for (int c = 0; c < CPL; ++c) vals[c] = smem[r * T + c * 32 + lane];
    float mx = -1e30f;
#pragma unroll
    for (int c = 0; c < CPL; ++c) mx = fmaxf(mx, vals[c]);
#pragma unroll
    for (int off = 16; off > 0; off >>= 1) mx = fmaxf(mx, __shfl_xor(mx, off, 32));
    float sum = 0.f;
#pragma unroll
    for (int c = 0; c < CPL; ++c) { vals[c] = __expf(vals[c] - mx); sum += vals[c]; }
#pragma unroll
    for (int off = 16; off > 0; off >>= 1) sum += __shfl_xor(sum, off, 32);
    const float inv = 1.f / sum;
    __syncthreads();                        // all reads of this row done before alias write
#pragma unroll
    for (int c = 0; c < CPL; ++c) {
      const float p = vals[c] * inv;
      const int col = c * 32 + lane;
      probs[r * T + col] = bf(p);
      if (emit_attn)                        // attn_last_T[b, key, query]
        attn_out[((size_t)b * T + col) * T + qt * 16 + r] = p;
    }
    __syncthreads();
  }

  // ---- upd = P @ V using transposed V (contiguous B-operand loads)
  v8f uacc[4] = {v8f{}, v8f{}, v8f{}, v8f{}};
  const unsigned short* VTb = VT + ((size_t)b * DM + h * DH) * T;
  for (int kc = 0; kc < T / 32; ++kc) {
    ABf pa;
    const int base = am * T + kc * 32;
    pa.u4[0] = *(const u32x4*)(probs + base + akb);
    pa.u4[1] = *(const u32x4*)(probs + base + 8 + akb);
#pragma unroll
    for (int g = 0; g < 4; ++g) {
      ABf vb;
      const unsigned short* vr = VTb + (size_t)(g * 16 + bn) * T + kc * 32 + bkb;
      vb.u4[0] = *(const u32x4*)(vr);
      vb.u4[1] = *(const u32x4*)(vr + 8);
      uacc[g] = wmma_bf16(pa, vb, uacc[g]);
    }
  }
  float* ur = upd + ((size_t)b * T + qt * 16) * DM + h * DH;
#pragma unroll
  for (int g = 0; g < 4; ++g)
#pragma unroll
    for (int r = 0; r < 8; ++r)
      ur[(size_t)(cm + r) * DM + g * 16 + cn] = uacc[g][r];
}

// --------------------------------------------- output GEMM (bf16 A, bf16 B)
// 64x64 block per wave. out[row,j] = XU16[row,:] @ Wo[j,:] + bo[j]
__global__ void __launch_bounds__(256)
outproj_kernel(const unsigned short* __restrict__ XU16,
               const unsigned short* __restrict__ W16,
               const float* __restrict__ bias, float* __restrict__ out) {
  const int lane = threadIdx.x & 31;
  const int wave = threadIdx.x >> 5;
  const int tileId = blockIdx.x * 8 + wave;
  const int Ntiles = DM / 64;                // 8
  const int tm = tileId / Ntiles, tn = tileId % Ntiles;

  const int am  = lane & 15;
  const int akb = (lane >> 4) * 16;
  const int bn  = lane & 15;
  const int bkb = (lane >> 4) * 16;
  const int rm  = tm * 64;
  const int jn  = tn * 64;

  const unsigned short* xbase = XU16 + ((size_t)(rm + am)) * DM;
  const unsigned short* wbase = W16 + ((size_t)(jn + bn)) * DM;

  v8f acc[4][4] = {};
  for (int kc = 0; kc < DM / 32; ++kc) {
    const int k0 = kc * 32 + akb;
    const int k1 = kc * 32 + bkb;
    ABf a[4], b[4];
#pragma unroll
    for (int i = 0; i < 4; ++i) {
      const unsigned short* xr = xbase + (size_t)i * 16 * DM;
      a[i].u4[0] = *(const u32x4*)(xr + k0);
      a[i].u4[1] = *(const u32x4*)(xr + k0 + 8);
    }
#pragma unroll
    for (int g = 0; g < 4; ++g) {
      const unsigned short* wr = wbase + (size_t)g * 16 * DM;
      b[g].u4[0] = *(const u32x4*)(wr + k1);
      b[g].u4[1] = *(const u32x4*)(wr + k1 + 8);
    }
#pragma unroll
    for (int i = 0; i < 4; ++i)
#pragma unroll
      for (int g = 0; g < 4; ++g)
        acc[i][g] = wmma_bf16(a[i], b[g], acc[i][g]);
  }

  const int cm = (lane >> 4) * 8;
  const int cn = lane & 15;
#pragma unroll
  for (int g = 0; g < 4; ++g) {
    const int jc = jn + g * 16 + cn;
    const float bj = bias[jc];
#pragma unroll
    for (int i = 0; i < 4; ++i)
#pragma unroll
      for (int r = 0; r < 8; ++r)
        out[(size_t)(rm + i * 16 + cm + r) * DM + jc] = acc[i][g][r] + bj;
  }
}

} // namespace

extern "C" void kernel_launch(void* const* d_in, const int* in_sizes, int n_in,
                              void* d_out, int out_size, void* d_ws, size_t ws_size,
                              hipStream_t stream) {
  (void)in_sizes; (void)n_in; (void)out_size; (void)ws_size;
  const float* v       = (const float*)d_in[0];
  const float* q       = (const float*)d_in[1];
  const float* v_mask  = (const float*)d_in[2];
  const float* q_mask  = (const float*)d_in[3];
  const float* q_trans = (const float*)d_in[4];
  const float* Wv4q    = (const float*)d_in[5];
  const float* bv4q    = (const float*)d_in[6];
  const float* Wq4v    = (const float*)d_in[7];
  const float* bq4v    = (const float*)d_in[8];
  const float* Wv      = (const float*)d_in[9];
  const float* bv      = (const float*)d_in[10];
  const float* Wq      = (const float*)d_in[11];
  const float* bq      = (const float*)d_in[12];
  const float* Wvo     = (const float*)d_in[13];
  const float* bvo     = (const float*)d_in[14];
  const float* Wqo     = (const float*)d_in[15];
  const float* bqo     = (const float*)d_in[16];

  float* out_v    = (float*)d_out;
  float* out_q    = out_v + (size_t)Bb * NOBJ * DM;
  float* out_attn = out_q + (size_t)Bb * MAXLEN * DM;

  char* ws = (char*)d_ws;
  size_t off = 0;
  auto alloc = [&](size_t bytes) -> void* {
    void* p = ws + off;
    off += (bytes + 255) & ~(size_t)255;
    return p;
  };
  unsigned short* Wv16  = (unsigned short*)alloc((size_t)3 * DM * DM * 2);
  unsigned short* Wq16  = (unsigned short*)alloc((size_t)3 * DM * DM * 2);
  unsigned short* Wvo16 = (unsigned short*)alloc((size_t)DM * DM * 2);
  unsigned short* Wqo16 = (unsigned short*)alloc((size_t)DM * DM * 2);
  float* vmean = (float*)alloc((size_t)Bb * DM * 4);
  float* gv    = (float*)alloc((size_t)Bb * DM * 4);
  float* gq    = (float*)alloc((size_t)Bb * DM * 4);
  unsigned short* v16  = (unsigned short*)alloc((size_t)Bb * NOBJ * DM * 2);
  unsigned short* q16  = (unsigned short*)alloc((size_t)Bb * MAXLEN * DM * 2);
  unsigned short* vKQ  = (unsigned short*)alloc((size_t)Bb * NOBJ * 1024 * 2);
  unsigned short* vVT  = (unsigned short*)alloc((size_t)Bb * DM * NOBJ * 2);
  unsigned short* qKQ  = (unsigned short*)alloc((size_t)Bb * MAXLEN * 1024 * 2);
  unsigned short* qVT  = (unsigned short*)alloc((size_t)Bb * DM * MAXLEN * 2);
  float* updV = (float*)alloc((size_t)Bb * NOBJ * DM * 4);
  float* updQ = (float*)alloc((size_t)Bb * MAXLEN * DM * 4);
  unsigned short* xuV16 = (unsigned short*)alloc((size_t)Bb * NOBJ * DM * 2);
  unsigned short* xuQ16 = (unsigned short*)alloc((size_t)Bb * MAXLEN * DM * 2);

  mean_kernel<<<Bb, 256, 0, stream>>>(v, v_mask, vmean);
  gate_kernel<<<Bb, 256, 0, stream>>>(vmean, q_trans, Wv4q, bv4q, Wq4v, bq4v, gq, gv);

  int n = 3 * DM * DM;
  cvt_bf16_kernel<<<(n / 8 + 255) / 256, 256, 0, stream>>>(Wv, Wv16, n);
  cvt_bf16_kernel<<<(n / 8 + 255) / 256, 256, 0, stream>>>(Wq, Wq16, n);
  n = DM * DM;
  cvt_bf16_kernel<<<(n / 8 + 255) / 256, 256, 0, stream>>>(Wvo, Wvo16, n);
  cvt_bf16_kernel<<<(n / 8 + 255) / 256, 256, 0, stream>>>(Wqo, Wqo16, n);
  int nv = Bb * NOBJ * DM;
  cvt_bf16_kernel<<<(nv / 8 + 255) / 256, 256, 0, stream>>>(v, v16, nv);
  int nq = Bb * MAXLEN * DM;
  cvt_bf16_kernel<<<(nq / 8 + 255) / 256, 256, 0, stream>>>(q, q16, nq);

  int tilesV = (Bb * NOBJ / 64) * (3 * DM / 64);       // 768*24 = 18432
  xform_kernel<NOBJ><<<tilesV / 8, 256, 0, stream>>>(v16, Wv16, bv, v_mask, gv, vKQ, vVT);
  int tilesQ = (Bb * MAXLEN / 64) * (3 * DM / 64);     // 256*24 = 6144
  xform_kernel<MAXLEN><<<tilesQ / 8, 256, 0, stream>>>(q16, Wq16, bq, q_mask, gq, qKQ, qVT);

  attn_kernel<NOBJ><<<Bb * H * (NOBJ / 16), 32, 0, stream>>>(vKQ, vVT, v_mask, updV, out_attn);
  attn_kernel<MAXLEN><<<Bb * H * (MAXLEN / 16), 32, 0, stream>>>(qKQ, qVT, q_mask, updQ, nullptr);

  addcvt_kernel<<<(nv / 8 + 255) / 256, 256, 0, stream>>>(v, updV, xuV16, nv);
  addcvt_kernel<<<(nq / 8 + 255) / 256, 256, 0, stream>>>(q, updQ, xuQ16, nq);

  outproj_kernel<<<(Bb * NOBJ / 64) * (DM / 64) / 8, 256, 0, stream>>>(xuV16, Wvo16, bvo, out_v);
  outproj_kernel<<<(Bb * MAXLEN / 64) * (DM / 64) / 8, 256, 0, stream>>>(xuQ16, Wqo16, bqo, out_q);
}
